// NodeEmbeddingLayer_19559281066587
// MI455X (gfx1250) — matrix operand order repaired and checked by reading, not compiled, not run on hardware
//
#include <hip/hip_runtime.h>
#include <hip/hip_bf16.h>

typedef __attribute__((ext_vector_type(16))) __bf16 v16bf;
typedef __attribute__((ext_vector_type(8)))  float  v8f;

#define NF    64
#define NH    128
#define NO    64
#define NCOEF 5
#define KANK  384   /* 64 (base path) + 64*5 (spline path) */

/* workspace layout (bytes) */
#define WC_PACK_OFF   0        /* 16 frags * 1KB */
#define WN_PACK_OFF   16384    /* 16 frags */
#define WU_PACK_OFF   32768    /* 16 frags */
#define KAN_PACK_OFF  49152    /* 48 frags */
#define SEG_OFF       131072   /* N*128 f32, then deg N f32 */

__device__ __forceinline__ v8f wmma_bf16(v16bf a, v16bf b, v8f c) {
  return __builtin_amdgcn_wmma_f32_16x16x32_bf16(false, a, false, b, (short)0, c,
                                                 false, false);
}

/* A-fragment (16x32 bf16) from an fp32 row: lane(<16)->M=lane K in {0..7,16..23},
   lane(>=16)->M=lane-16, K in {8..15,24..31}; koff = (lane>=16)?8:0 */
__device__ __forceinline__ v16bf load_a_f32(const float* __restrict__ row, int k0, int koff) {
  const float* p0 = row + k0 + koff;
  const float* p1 = p0 + 16;
  v16bf a;
#pragma unroll
  for (int i = 0; i < 8; ++i) { a[i] = (__bf16)p0[i]; a[i + 8] = (__bf16)p1[i]; }
  return a;
}

__device__ __forceinline__ v16bf load_a_bf16(const __bf16* row, int k0, int koff) {
  const __bf16* p0 = row + k0 + koff;
  const __bf16* p1 = p0 + 16;
  v16bf a;
#pragma unroll
  for (int i = 0; i < 8; ++i) { a[i] = p0[i]; a[i + 8] = p1[i]; }
  return a;
}

/* quadratic B-spline bases on uniform knots t_j = -1 + (j-2)*2/3, j=0..7 */
__device__ __forceinline__ void bspline5(float u, float* b2) {
  const float h = 2.0f / 3.0f;
  float t[8];
#pragma unroll
  for (int j = 0; j < 8; ++j) t[j] = -1.0f + (float)(j - 2) * h;
  float b0[7];
#pragma unroll
  for (int j = 0; j < 7; ++j) b0[j] = (u >= t[j] && u < t[j + 1]) ? 1.0f : 0.0f;
  float b1[6];
#pragma unroll
  for (int j = 0; j < 6; ++j)
    b1[j] = (u - t[j]) / h * b0[j] + (t[j + 2] - u) / h * b0[j + 1];
#pragma unroll
  for (int j = 0; j < 5; ++j)
    b2[j] = (u - t[j]) / (2.0f * h) * b1[j] + (t[j + 3] - u) / (2.0f * h) * b1[j + 1];
}

/* ---- pack all weights into bf16 B-fragment layout ------------------------ */
__global__ void pack_kernel(const float* __restrict__ Wc, const float* __restrict__ Wn,
                            const float* __restrict__ Wu, const float* __restrict__ base_w,
                            const float* __restrict__ spline_w, const float* __restrict__ scaler,
                            __bf16* __restrict__ pack) {
  int b = blockIdx.x;              /* fragment id 0..95 */
  int tid = threadIdx.x;           /* 512 threads = 32 lanes x 16 elems   */
  int lane = tid >> 4, e = tid & 15;
  int khalf = (lane >= 16) ? 8 : 0;
  int kk = (e < 8) ? (khalf + e) : (16 + khalf + (e - 8));
  int nit = lane & 15;
  float val = 0.0f;
  if (b < 16) {                    /* Wc: [H,F] -> K=F(2 chunks), N=H(8 tiles) */
    int fb = b, c = fb >> 3, t = fb & 7;
    int k = c * 32 + kk, n = t * 16 + nit;
    val = Wc[n * NF + k];
  } else if (b < 32) {             /* Wn */
    int fb = b - 16, c = fb >> 3, t = fb & 7;
    int k = c * 32 + kk, n = t * 16 + nit;
    val = Wn[n * NF + k];
  } else if (b < 48) {             /* Wu: [O,H] -> K=H(4 chunks), N=O(4 tiles) */
    int fb = b - 32, c = fb >> 2, t = fb & 3;
    int k = c * 32 + kk, n = t * 16 + nit;
    val = Wu[n * NH + k];
  } else {                         /* fused KAN weight: K=384(12), N=O(4) */
    int fb = b - 48, c = fb >> 2, t = fb & 3;
    int k = c * 32 + kk, n = t * 16 + nit;
    if (k < NO) {
      val = base_w[n * NO + k];
    } else {
      int j = k - NO, i = j / NCOEF, q = j - NCOEF * i;
      val = spline_w[(n * NO + i) * NCOEF + q] * scaler[n * NO + i];
    }
  }
  pack[(size_t)b * 512 + lane * 16 + e] = (__bf16)val;
}

/* ---- degree via run-merged atomics (seg_ids sorted) ---------------------- */
__global__ void deg_kernel(const int* __restrict__ seg, float* __restrict__ deg, int E) {
  int t = blockIdx.x * blockDim.x + threadIdx.x;
  int start = t * 32;
  if (start >= E) return;
  int end = min(start + 32, E);
  int cur = seg[start];
  float cnt = 1.0f;
  for (int i = start + 1; i < end; ++i) {
    int s = seg[i];
    if (s == cur) cnt += 1.0f;
    else { atomicAdd(&deg[cur], cnt); cur = s; cnt = 1.0f; }
  }
  atomicAdd(&deg[cur], cnt);
}

/* ---- edge GEMM + attention + segment-sum scatter ------------------------- */
__global__ void __launch_bounds__(256)
edge_kernel(const float* __restrict__ contexts, const float* __restrict__ attn,
            const float* __restrict__ bc, const int* __restrict__ seg,
            const v16bf* __restrict__ wc_pack, float* __restrict__ seg_sum, int E) {
  int wv = threadIdx.x >> 5, lane = threadIdx.x & 31;
  int tile = blockIdx.x * 8 + wv;
  int base = tile * 16;
  if (base >= E) return;
  int mrow = lane & 15;
  int koff = (lane >= 16) ? 8 : 0; /* also the D-matrix M offset */

  const float* arow = contexts + (size_t)(base + mrow) * NF;
  v16bf a0 = load_a_f32(arow, 0, koff);
  v16bf a1 = load_a_f32(arow, 32, koff);

  int   segv[8];
  float attnv[8];
#pragma unroll
  for (int r = 0; r < 8; ++r) {
    int e = base + koff + r;
    segv[r]  = seg[e];
    attnv[r] = attn[e];
  }

#pragma unroll
  for (int t = 0; t < 8; ++t) {
    v8f acc = {};
    v16bf b0 = wc_pack[(size_t)(0 * 8 + t) * 32 + lane];
    v16bf b1 = wc_pack[(size_t)(1 * 8 + t) * 32 + lane];
    acc = wmma_bf16(a0, b0, acc);
    acc = wmma_bf16(a1, b1, acc);
    int h = t * 16 + mrow;
    float bch = bc[h];
    /* run-merged scatter: D row r is edge base+koff+r, contiguous in r */
    float run = (acc[0] + bch) * attnv[0];
    int cur = segv[0];
#pragma unroll
    for (int r = 1; r < 8; ++r) {
      float v = (acc[r] + bch) * attnv[r];
      if (segv[r] == cur) run += v;
      else {
        atomicAdd(&seg_sum[(size_t)cur * NH + h], run);
        cur = segv[r]; run = v;
      }
    }
    atomicAdd(&seg_sum[(size_t)cur * NH + h], run);
  }
}

/* ---- node pipeline: GEMM1 -> GEMM2 -> KAN GEMM --------------------------- */
__global__ void __launch_bounds__(128)
node_kernel(const float* __restrict__ x, const float* __restrict__ bn,
            const float* __restrict__ bu, const float* __restrict__ seg_sum,
            const float* __restrict__ deg, const v16bf* __restrict__ wn_pack,
            const v16bf* __restrict__ wu_pack, const v16bf* __restrict__ kan_pack,
            float* __restrict__ out, int n) {
  __shared__ __align__(16) unsigned char smem[4 * 12288]; /* 12KB per wave */
  int wv = threadIdx.x >> 5, lane = threadIdx.x & 31;
  int tile = blockIdx.x * 4 + wv;
  int node0 = tile * 16;
  if (node0 >= n) return;
  int mrow = lane & 15;
  int koff = (lane >= 16) ? 8 : 0;

  float* hid  = (float*)(smem + wv * 12288);   /* [16][128] f32  (8KB)  */
  __bf16* kan = (__bf16*)(smem + wv * 12288);  /* [16][384] bf16 (12KB) */

  /* per-lane node metadata for its 8 D rows (M = koff + r) */
  float invdeg[8];
  bool  hasctx[8];
#pragma unroll
  for (int r = 0; r < 8; ++r) {
    float d = deg[node0 + koff + r];
    hasctx[r] = d > 0.0f;
    invdeg[r] = 1.0f / fmaxf(d, 1.0f);
  }

  /* stage 1: hid = x@Wn^T + bn + seg_sum/deg  -> LDS fp32 */
  const float* xrow = x + (size_t)(node0 + mrow) * NF;
  v16bf a0 = load_a_f32(xrow, 0, koff);
  v16bf a1 = load_a_f32(xrow, 32, koff);
#pragma unroll
  for (int t = 0; t < 8; ++t) {
    v8f acc = {};
    acc = wmma_bf16(a0, wn_pack[(size_t)(0 * 8 + t) * 32 + lane], acc);
    acc = wmma_bf16(a1, wn_pack[(size_t)(1 * 8 + t) * 32 + lane], acc);
    int h = t * 16 + mrow;
    float bnh = bn[h];
#pragma unroll
    for (int r = 0; r < 8; ++r) {
      int node = node0 + koff + r;
      float v = acc[r] + bnh + seg_sum[(size_t)node * NH + h] * invdeg[r];
      hid[(koff + r) * NH + h] = v;
    }
  }
  asm volatile("s_wait_dscnt 0" ::: "memory");

  /* stage 2: upd = hid@Wu^T (+bu later), K=128 */
  v8f uacc[4];
#pragma unroll
  for (int nt = 0; nt < 4; ++nt) uacc[nt] = (v8f){};
  const float* hrow = hid + mrow * NH;
#pragma unroll
  for (int c = 0; c < 4; ++c) {
    v16bf a = load_a_f32(hrow, c * 32, koff);
#pragma unroll
    for (int nt = 0; nt < 4; ++nt)
      uacc[nt] = wmma_bf16(a, wu_pack[(size_t)(c * 4 + nt) * 32 + lane], uacc[nt]);
  }

  /* stage 3: silu + b-spline bases -> LDS bf16 A-matrix [16][384] */
#pragma unroll
  for (int nt = 0; nt < 4; ++nt) {
    int o = nt * 16 + mrow;
    float buo = bu[o];
#pragma unroll
    for (int r = 0; r < 8; ++r) {
      int M = koff + r;
      float u = uacc[nt][r] + buo;
      float sl = u / (1.0f + __expf(-u));
      float b2[5];
      bspline5(u, b2);
      __bf16* row = kan + M * KANK;
      row[o] = (__bf16)sl;
#pragma unroll
      for (int q = 0; q < NCOEF; ++q) row[NO + o * NCOEF + q] = (__bf16)b2[q];
    }
  }
  asm volatile("s_wait_dscnt 0" ::: "memory");

  /* stage 4: out = [silu|bases] @ fused KAN weight, K=384 */
  v8f oacc[4];
#pragma unroll
  for (int nt = 0; nt < 4; ++nt) oacc[nt] = (v8f){};
  const __bf16* krow = kan + mrow * KANK;
#pragma unroll
  for (int c = 0; c < 12; ++c) {
    v16bf a = load_a_bf16(krow, c * 32, koff);
#pragma unroll
    for (int nt = 0; nt < 4; ++nt)
      oacc[nt] = wmma_bf16(a, kan_pack[(size_t)(c * 4 + nt) * 32 + lane], oacc[nt]);
  }

  /* epilogue: where(deg>0, kan_out, x)  (F == O == 64) */
#pragma unroll
  for (int nt = 0; nt < 4; ++nt) {
    int o = nt * 16 + mrow;
#pragma unroll
    for (int r = 0; r < 8; ++r) {
      int node = node0 + koff + r;
      float v = oacc[nt][r];
      if (!hasctx[r]) v = x[(size_t)node * NF + o];
      out[(size_t)node * NO + o] = v;
    }
  }
}

extern "C" void kernel_launch(void* const* d_in, const int* in_sizes, int n_in,
                              void* d_out, int out_size, void* d_ws, size_t ws_size,
                              hipStream_t stream) {
  (void)n_in; (void)out_size; (void)ws_size;
  const float* x        = (const float*)d_in[0];
  const float* contexts = (const float*)d_in[1];
  const float* attn     = (const float*)d_in[2];
  const int*   seg      = (const int*)d_in[3];
  const float* Wn       = (const float*)d_in[4];
  const float* bn       = (const float*)d_in[5];
  const float* Wc       = (const float*)d_in[6];
  const float* bc       = (const float*)d_in[7];
  const float* Wu       = (const float*)d_in[8];
  const float* bu       = (const float*)d_in[9];
  const float* base_w   = (const float*)d_in[10];
  const float* spline_w = (const float*)d_in[11];
  const float* scaler   = (const float*)d_in[12];

  int E = in_sizes[2];          /* attn is [E] */
  int N = in_sizes[0] / NF;     /* x is [N,F]  */

  char*  ws      = (char*)d_ws;
  float* seg_sum = (float*)(ws + SEG_OFF);
  float* deg     = (float*)(ws + SEG_OFF + (size_t)N * NH * 4);

  /* zero seg_sum + deg */
  hipMemsetAsync(ws + SEG_OFF, 0, (size_t)N * NH * 4 + (size_t)N * 4, stream);

  pack_kernel<<<96, 512, 0, stream>>>(Wc, Wn, Wu, base_w, spline_w, scaler, (__bf16*)ws);

  int degThreads = (E + 31) / 32;
  deg_kernel<<<(degThreads + 255) / 256, 256, 0, stream>>>(seg, deg, E);

  int etiles = (E + 15) / 16;
  edge_kernel<<<(etiles + 7) / 8, 256, 0, stream>>>(
      contexts, attn, bc, seg, (const v16bf*)(ws + WC_PACK_OFF), seg_sum, E);

  int ntiles = (N + 15) / 16;
  node_kernel<<<(ntiles + 3) / 4, 128, 0, stream>>>(
      x, bn, bu, seg_sum, deg,
      (const v16bf*)(ws + WN_PACK_OFF),
      (const v16bf*)(ws + WU_PACK_OFF),
      (const v16bf*)(ws + KAN_PACK_OFF),
      (float*)d_out, N);
}
